// PointNetEncoder_62242666053807
// MI455X (gfx1250) — compile-verified, matrix operand order checked by reading, not run on hardware
//
#include <hip/hip_runtime.h>
#include <hip/hip_bf16.h>

#define EPSV   1e-6f
#define BN_EPS 1e-5f

#define B_  16
#define N_  2048
#define K_  16
#define C1_ 21
#define C2_ 42
#define C3_ 341
#define C6_ 682
#define Cq_ 170
#define NC_ (3 * N_)        // 6144 columns per batch (xyz,n flattened)
#define NT_ (NC_ / 16)      // 384 column tiles
#define NK_ (N_ * K_)       // 32768

typedef _Float16 v8h  __attribute__((ext_vector_type(8)));
typedef _Float16 v16h __attribute__((ext_vector_type(16)));
typedef float    v8f  __attribute__((ext_vector_type(8)));

static inline int divup(int a, int b) { return (a + b - 1) / b; }

// fragment slot (i in 0..15) -> K offset within a 32-row panel, given lane half
__device__ __host__ inline int frag_k(int i, int half) {
    int kb = half * 8;
    return (i < 8) ? (kb + i) : (16 + kb + (i - 8));
}

// ---------------------------------------------------------------- kNN top-16
__global__ void knn_kernel(const float* __restrict__ x, int* __restrict__ idx) {
    __shared__ float sx[N_], sy[N_], sz[N_];
    const int b = blockIdx.y;
    const float* xb = x + (size_t)b * N_ * 3;
    for (int i = threadIdx.x; i < N_; i += blockDim.x) {
        sx[i] = xb[i * 3 + 0]; sy[i] = xb[i * 3 + 1]; sz[i] = xb[i * 3 + 2];
    }
    __syncthreads();
    const int n = blockIdx.x * blockDim.x + threadIdx.x;
    const float cx = sx[n], cy = sy[n], cz = sz[n];
    float best[K_]; int bi[K_];
#pragma unroll
    for (int k = 0; k < K_; ++k) { best[k] = -1e30f; bi[k] = 0; }
    for (int m = 0; m < N_; ++m) {
        float dx = cx - sx[m], dy = cy - sy[m], dz = cz - sz[m];
        float v = -(dx * dx + dy * dy + dz * dz);   // negative sq dist
        if (v > best[K_ - 1]) {
            int p = K_ - 1;
            while (p > 0 && best[p - 1] < v) { best[p] = best[p - 1]; bi[p] = bi[p - 1]; --p; }
            best[p] = v; bi[p] = m;
        }
    }
    int* ob = idx + (size_t)b * NK_ + (size_t)n * K_;
#pragma unroll
    for (int k = 0; k < K_; ++k) ob[k] = bi[k];
}

__global__ void zero_kernel(float* p, int n) {
    int t = blockIdx.x * blockDim.x + threadIdx.x;
    if (t < n) p[t] = 0.f;
}

// ------------------------------------------------ edge features (3 VN chans)
__device__ inline void edge_feat(const float* xb, const int* ib, int n, int k,
                                 float v[3][3]) {
    float cx = xb[n * 3 + 0], cy = xb[n * 3 + 1], cz = xb[n * 3 + 2];
    int m = ib[n * K_ + k];
    float nx = xb[m * 3 + 0], ny = xb[m * 3 + 1], nz = xb[m * 3 + 2];
    v[0][0] = nx - cx; v[0][1] = ny - cy; v[0][2] = nz - cz;   // nbr - ctr
    v[1][0] = cx;      v[1][1] = cy;      v[1][2] = cz;        // ctr
    v[2][0] = ny * cz - nz * cy;                               // nbr x ctr
    v[2][1] = nz * cx - nx * cz;
    v[2][2] = nx * cy - ny * cx;
}

// pass 1: batch statistics of ||wf x feat|| per channel (atomics into stats)
__global__ void edge_stats_kernel(const float* __restrict__ x, const int* __restrict__ idx,
                                  const float* __restrict__ wf, float* __restrict__ stats) {
    __shared__ float ssum[C1_], ssq[C1_];
    if (threadIdx.x < C1_) { ssum[threadIdx.x] = 0.f; ssq[threadIdx.x] = 0.f; }
    __syncthreads();
    int t = blockIdx.x * blockDim.x + threadIdx.x;          // exactly B*N*K threads
    int b = t / NK_, r = t % NK_, n = r / K_, k = r % K_;
    const float* xb = x + (size_t)b * N_ * 3;
    const int* ib = idx + (size_t)b * NK_;
    float v[3][3];
    edge_feat(xb, ib, n, k, v);
    for (int c = 0; c < C1_; ++c) {
        float w0 = wf[c * 3 + 0], w1 = wf[c * 3 + 1], w2 = wf[c * 3 + 2];
        float px = w0 * v[0][0] + w1 * v[1][0] + w2 * v[2][0];
        float py = w0 * v[0][1] + w1 * v[1][1] + w2 * v[2][1];
        float pz = w0 * v[0][2] + w1 * v[1][2] + w2 * v[2][2];
        float nrm = sqrtf(px * px + py * py + pz * pz) + EPSV;
        atomicAdd(&ssum[c], nrm);
        atomicAdd(&ssq[c], nrm * nrm);
    }
    __syncthreads();
    if (threadIdx.x < C1_) {
        atomicAdd(&stats[threadIdx.x], ssum[threadIdx.x]);
        atomicAdd(&stats[C1_ + threadIdx.x], ssq[threadIdx.x]);
    }
}

// pass 2: VN-BN + leaky-relu(0) + pool-direction dot per (b,n,k)
__global__ void edge_apply_kernel(const float* __restrict__ x, const int* __restrict__ idx,
                                  const float* __restrict__ wf, const float* __restrict__ wd,
                                  const float* __restrict__ g, const float* __restrict__ bt,
                                  const float* __restrict__ stats, const float* __restrict__ pwd,
                                  float* __restrict__ hE, float* __restrict__ dotE) {
    int t = blockIdx.x * blockDim.x + threadIdx.x;
    int b = t / NK_, r = t % NK_, n = r / K_, k = r % K_;
    const float* xb = x + (size_t)b * N_ * 3;
    const int* ib = idx + (size_t)b * NK_;
    float v[3][3];
    edge_feat(xb, ib, n, k, v);
    const float inv_cnt = 1.0f / (float)(B_ * N_ * K_);
    float hv[C1_][3];
    for (int c = 0; c < C1_; ++c) {
        float w0 = wf[c * 3 + 0], w1 = wf[c * 3 + 1], w2 = wf[c * 3 + 2];
        float px = w0 * v[0][0] + w1 * v[1][0] + w2 * v[2][0];
        float py = w0 * v[0][1] + w1 * v[1][1] + w2 * v[2][1];
        float pz = w0 * v[0][2] + w1 * v[1][2] + w2 * v[2][2];
        float nrm = sqrtf(px * px + py * py + pz * pz) + EPSV;
        float mean = stats[c] * inv_cnt;
        float var = stats[C1_ + c] * inv_cnt - mean * mean;
        float nbn = (nrm - mean) * rsqrtf(var + BN_EPS) * g[c] + bt[c];
        float s = nbn / nrm;
        px *= s; py *= s; pz *= s;
        float d0 = wd[c * 3 + 0], d1 = wd[c * 3 + 1], d2 = wd[c * 3 + 2];
        float dx = d0 * v[0][0] + d1 * v[1][0] + d2 * v[2][0];
        float dy = d0 * v[0][1] + d1 * v[1][1] + d2 * v[2][1];
        float dz = d0 * v[0][2] + d1 * v[1][2] + d2 * v[2][2];
        float dot = px * dx + py * dy + pz * dz;
        if (dot < 0.f) {
            float f = dot / (dx * dx + dy * dy + dz * dz + EPSV);
            px -= f * dx; py -= f * dy; pz -= f * dz;
        }
        hv[c][0] = px; hv[c][1] = py; hv[c][2] = pz;
    }
    for (int c = 0; c < C1_; ++c) {
        float dpx = 0.f, dpy = 0.f, dpz = 0.f;
        for (int j = 0; j < C1_; ++j) {
            float w = pwd[c * C1_ + j];
            dpx += w * hv[j][0]; dpy += w * hv[j][1]; dpz += w * hv[j][2];
        }
        float dotc = dpx * hv[c][0] + dpy * hv[c][1] + dpz * hv[c][2];
        size_t hb = (((size_t)b * C1_ + c) * 3) * NK_ + (size_t)n * K_ + k;
        hE[hb] = hv[c][0]; hE[hb + NK_] = hv[c][1]; hE[hb + 2 * NK_] = hv[c][2];
        dotE[((size_t)b * C1_ + c) * NK_ + (size_t)n * K_ + k] = dotc;
    }
}

__global__ void pool_kernel(const float* __restrict__ hE, const float* __restrict__ dotE,
                            float* __restrict__ h0) {
    int t = blockIdx.x * blockDim.x + threadIdx.x;      // < B*C1*N
    if (t >= B_ * C1_ * N_) return;
    int n = t % N_, bc = t / N_;
    size_t db = (size_t)bc * NK_ + (size_t)n * K_;
    int km = 0; float bv = dotE[db];
#pragma unroll
    for (int k = 1; k < K_; ++k) { float v = dotE[db + k]; if (v > bv) { bv = v; km = k; } }
    size_t hb = (size_t)bc * 3 * NK_ + (size_t)n * K_ + km;
    size_t ob = (size_t)bc * 3 * N_ + n;
    h0[ob] = hE[hb]; h0[ob + N_] = hE[hb + NK_]; h0[ob + 2 * N_] = hE[hb + 2 * NK_];
}

// ------------------------- small dual linear (wf & wd), weights in LDS (VALU)
__global__ void dual_linear_kernel(const float* __restrict__ in,
                                   const float* __restrict__ wf, const float* __restrict__ wd,
                                   float* __restrict__ P, float* __restrict__ D,
                                   int Cin, int Cout) {
    __shared__ float swf[C2_ * C2_], swd[C2_ * C2_];
    int nw = Cin * Cout;
    for (int i = threadIdx.x; i < nw; i += blockDim.x) { swf[i] = wf[i]; swd[i] = wd[i]; }
    __syncthreads();
    int t = blockIdx.x * blockDim.x + threadIdx.x;      // exactly B*NC threads
    int b = t / NC_, col = t % NC_;
    float xv[C2_];
    const float* ib = in + (size_t)b * Cin * NC_ + col;
    for (int i = 0; i < Cin; ++i) xv[i] = ib[(size_t)i * NC_];
    float* Pb = P + (size_t)b * Cout * NC_ + col;
    float* Db = D + (size_t)b * Cout * NC_ + col;
    for (int c = 0; c < Cout; ++c) {
        float af = 0.f, ad = 0.f;
        for (int i = 0; i < Cin; ++i) { af += swf[c * Cin + i] * xv[i]; ad += swd[c * Cin + i] * xv[i]; }
        Pb[(size_t)c * NC_] = af; Db[(size_t)c * NC_] = ad;
    }
}

// ------------------------------------------ VN-BN stats (one block / channel)
__global__ void bn_stats_kernel(const float* __restrict__ P, float* __restrict__ stats, int C) {
    __shared__ float rs[256], rq[256];
    int c = blockIdx.x;
    float s = 0.f, q = 0.f;
    for (int i = threadIdx.x; i < B_ * N_; i += blockDim.x) {
        int b = i / N_, n = i % N_;
        size_t base = (((size_t)b * C + c) * 3) * N_ + n;
        float px = P[base], py = P[base + N_], pz = P[base + 2 * N_];
        float nrm = sqrtf(px * px + py * py + pz * pz) + EPSV;
        s += nrm; q += nrm * nrm;
    }
    rs[threadIdx.x] = s; rq[threadIdx.x] = q;
    __syncthreads();
    for (int off = blockDim.x / 2; off > 0; off >>= 1) {
        if ((int)threadIdx.x < off) { rs[threadIdx.x] += rs[threadIdx.x + off]; rq[threadIdx.x] += rq[threadIdx.x + off]; }
        __syncthreads();
    }
    if (threadIdx.x == 0) { stats[c] = rs[0]; stats[C + c] = rq[0]; }
}

// ------------------------ VN-BN apply (+ optional leaky-relu(0) projection)
__global__ void bn_apply_kernel(const float* __restrict__ P, const float* __restrict__ D,
                                const float* __restrict__ g, const float* __restrict__ bt,
                                const float* __restrict__ stats, float* __restrict__ OUT,
                                int C, float inv_cnt, int lrelu) {
    int t = blockIdx.x * blockDim.x + threadIdx.x;      // < B*C*N
    if (t >= B_ * C * N_) return;
    int n = t % N_, bc = t / N_, c = bc % C;
    size_t base = (size_t)bc * 3 * N_ + n;
    float px = P[base], py = P[base + N_], pz = P[base + 2 * N_];
    float nrm = sqrtf(px * px + py * py + pz * pz) + EPSV;
    float mean = stats[c] * inv_cnt;
    float var = stats[C + c] * inv_cnt - mean * mean;
    float nbn = (nrm - mean) * rsqrtf(var + BN_EPS) * g[c] + bt[c];
    float s = nbn / nrm;
    px *= s; py *= s; pz *= s;
    if (lrelu) {
        float dx = D[base], dy = D[base + N_], dz = D[base + 2 * N_];
        float dot = px * dx + py * dy + pz * dz;
        if (dot < 0.f) {
            float f = dot / (dx * dx + dy * dy + dz * dz + EPSV);
            px -= f * dx; py -= f * dy; pz -= f * dz;
        }
    }
    OUT[base] = px; OUT[base + N_] = py; OUT[base + 2 * N_] = pz;
}

// ---------------- pack f32 activations -> fragment-native f16 B panels
// Xp layout: (((b*KP + kp)*NT + nt)*32 + lane)*16 + i ; lane = half*16 + col%16
__global__ void pack_b_kernel(const float* __restrict__ src, _Float16* __restrict__ dst,
                              int C, int KP) {
    size_t t = (size_t)blockIdx.x * blockDim.x + threadIdx.x;
    size_t total = (size_t)B_ * KP * NT_ * 512;
    if (t >= total) return;
    int i    = (int)(t & 15);
    int lane = (int)((t >> 4) & 31);
    size_t rest = t >> 9;
    int nt = (int)(rest % NT_); rest /= NT_;
    int kp = (int)(rest % KP);
    int b  = (int)(rest / KP);
    int c = lane & 15, half = lane >> 4;
    int k = kp * 32 + frag_k(i, half);
    int col = nt * 16 + c;
    _Float16 v = (_Float16)0.f;
    if (k < C) v = (_Float16)src[((size_t)b * C + k) * NC_ + col];
    dst[t] = v;
}

// pack for the 682-channel concat input (rows 341..681 = broadcast h_mean)
__global__ void pack_b682_kernel(const float* __restrict__ h3, const float* __restrict__ hmean,
                                 _Float16* __restrict__ dst, int KP) {
    size_t t = (size_t)blockIdx.x * blockDim.x + threadIdx.x;
    size_t total = (size_t)B_ * KP * NT_ * 512;
    if (t >= total) return;
    int i    = (int)(t & 15);
    int lane = (int)((t >> 4) & 31);
    size_t rest = t >> 9;
    int nt = (int)(rest % NT_); rest /= NT_;
    int kp = (int)(rest % KP);
    int b  = (int)(rest / KP);
    int c = lane & 15, half = lane >> 4;
    int k = kp * 32 + frag_k(i, half);
    int col = nt * 16 + c;
    float v = 0.f;
    if (k < C3_)      v = h3[((size_t)b * C3_ + k) * NC_ + col];
    else if (k < C6_) v = hmean[((size_t)b * C3_ + (k - C3_)) * 3 + (col / N_)];
    dst[t] = (_Float16)v;
}

// ---------------- pack f32 weights -> fragment-native f16 A panels
// Wa layout: ((mt*KP + kp)*32 + lane)*16 + i ; lane = half*16 + row%16
__global__ void pack_a_kernel(const float* __restrict__ W, _Float16* __restrict__ dst,
                              int M, int Kc, int KP, int MT) {
    int t = blockIdx.x * blockDim.x + threadIdx.x;
    int total = MT * KP * 512;
    if (t >= total) return;
    int i    = t & 15;
    int lane = (t >> 4) & 31;
    int kp   = (t >> 9) % KP;
    int mt   = (t >> 9) / KP;
    int mr = lane & 15, half = lane >> 4;
    int k = kp * 32 + frag_k(i, half);
    int m = mt * 16 + mr;
    dst[t] = (m < M && k < Kc) ? (_Float16)W[m * Kc + k] : (_Float16)0.f;
}

// --------------------------------------------------------- WMMA batched GEMM
// OUT[b] (M x NC) = W (M x K) * X[b] (K x NC), fragment-native f16 operands.
// One wave per (16 rows x 64 cols) strip: A fragment reused across 4 WMMAs;
// steady state = 10x global_load_b128 + 4x v_wmma_f32_16x16x32_f16.
__global__ __launch_bounds__(32)
void wmma_gemm_kernel(const _Float16* __restrict__ Wa, const _Float16* __restrict__ Xp,
                      float* __restrict__ OUT, int M, int KP, int Nc) {
    const int lane = threadIdx.x;
    const int nt0  = blockIdx.x * 4;
    const int mt   = blockIdx.y;
    const int b    = blockIdx.z;
    const int half = lane >> 4;
    const int l16  = lane & 15;

    const _Float16* Ab = Wa + ((size_t)mt * KP) * 512 + lane * 16;
    const _Float16* Bb = Xp + (((size_t)b * KP) * NT_ + nt0) * 512 + lane * 16;
    const size_t bstride = (size_t)NT_ * 512;

    v8f acc0 = {}, acc1 = {}, acc2 = {}, acc3 = {};
    for (int kp = 0; kp < KP; ++kp) {
        v16h af = *(const v16h*)(Ab + (size_t)kp * 512);
        const _Float16* bp = Bb + (size_t)kp * bstride;
        v16h b0 = *(const v16h*)(bp);
        v16h b1 = *(const v16h*)(bp + 512);
        v16h b2 = *(const v16h*)(bp + 1024);
        v16h b3 = *(const v16h*)(bp + 1536);
        if (kp + 1 < KP) __builtin_prefetch(bp + bstride, 0, 1);
        acc0 = __builtin_amdgcn_wmma_f32_16x16x32_f16(false, af, false, b0, (short)0, acc0, false, false);
        acc1 = __builtin_amdgcn_wmma_f32_16x16x32_f16(false, af, false, b1, (short)0, acc1, false, false);
        acc2 = __builtin_amdgcn_wmma_f32_16x16x32_f16(false, af, false, b2, (short)0, acc2, false, false);
        acc3 = __builtin_amdgcn_wmma_f32_16x16x32_f16(false, af, false, b3, (short)0, acc3, false, false);
    }

    // D layout: VGPR v -> row mt*16 + v + 8*half, col = tile base + l16
    float* Ob = OUT + (size_t)b * M * Nc;
    const int rbase = mt * 16 + half * 8;
#pragma unroll
    for (int v = 0; v < 8; ++v) {
        int row = rbase + v;
        if (row < M) {
            float* orow = Ob + (size_t)row * Nc + nt0 * 16 + l16;
            orow[0]  = acc0[v];
            orow[16] = acc1[v];
            orow[32] = acc2[v];
            orow[48] = acc3[v];
        }
    }
}

// ------------------------------------------------- mean over N
__global__ void mean_kernel(const float* __restrict__ h3, float* __restrict__ hmean) {
    int t = blockIdx.x * blockDim.x + threadIdx.x;      // < B*C3*3
    if (t >= B_ * C3_ * 3) return;
    const float* p = h3 + (size_t)t * N_;
    float s = 0.f;
    for (int n = 0; n < N_; ++n) s += p[n];
    hmean[t] = s * (1.0f / (float)N_);
}

// --------------------------------------------------- last linear 170 -> 3x3
__global__ void sl_kernel(const float* __restrict__ z2, const float* __restrict__ w,
                          float* __restrict__ zfr) {
    __shared__ float sw[3 * Cq_];
    for (int i = threadIdx.x; i < 3 * Cq_; i += blockDim.x) sw[i] = w[i];
    __syncthreads();
    int t = blockIdx.x * blockDim.x + threadIdx.x;      // exactly B*NC
    int b = t / NC_, col = t % NC_;
    float a0 = 0.f, a1 = 0.f, a2 = 0.f;
    const float* zb = z2 + (size_t)b * Cq_ * NC_ + col;
    for (int i = 0; i < Cq_; ++i) {
        float xv = zb[(size_t)i * NC_];
        a0 += sw[0 * Cq_ + i] * xv; a1 += sw[1 * Cq_ + i] * xv; a2 += sw[2 * Cq_ + i] * xv;
    }
    float* ob = zfr + (size_t)b * 3 * NC_ + col;
    ob[0] = a0; ob[NC_] = a1; ob[2 * NC_] = a2;
}

// ----------------------------- frame einsum + max over N -> out (B, 2046)
__global__ void final_kernel(const float* __restrict__ h3, const float* __restrict__ hmean,
                             const float* __restrict__ zfr, float* __restrict__ out) {
    int t = blockIdx.x * blockDim.x + threadIdx.x;      // < B*682*3
    if (t >= B_ * C6_ * 3) return;
    int kxy = t % 3, bi = t / 3;
    int i = bi % C6_, b = bi / C6_;
    const float* zb = zfr + (size_t)b * 9 * N_;
    float m = -1e30f;
    if (i < C3_) {
        const float* hb = h3 + (((size_t)b * C3_ + i) * 3) * N_;
        for (int n = 0; n < N_; ++n) {
            float v = hb[n]          * zb[(0 * 3 + kxy) * N_ + n]
                    + hb[N_ + n]     * zb[(1 * 3 + kxy) * N_ + n]
                    + hb[2 * N_ + n] * zb[(2 * 3 + kxy) * N_ + n];
            m = fmaxf(m, v);
        }
    } else {
        const float* hm = hmean + ((size_t)b * C3_ + (i - C3_)) * 3;
        float h0v = hm[0], h1v = hm[1], h2v = hm[2];
        for (int n = 0; n < N_; ++n) {
            float v = h0v * zb[(0 * 3 + kxy) * N_ + n]
                    + h1v * zb[(1 * 3 + kxy) * N_ + n]
                    + h2v * zb[(2 * 3 + kxy) * N_ + n];
            m = fmaxf(m, v);
        }
    }
    out[(size_t)b * 2046 + i * 3 + kxy] = m;
}

// ============================================================================
extern "C" void kernel_launch(void* const* d_in, const int* in_sizes, int n_in,
                              void* d_out, int out_size, void* d_ws, size_t ws_size,
                              hipStream_t stream) {
    const float* x       = (const float*)d_in[0];
    const float* cp_wf   = (const float*)d_in[1];
    const float* cp_wd   = (const float*)d_in[2];
    const float* cp_g    = (const float*)d_in[3];
    const float* cp_b    = (const float*)d_in[4];
    const float* pool_wd = (const float*)d_in[5];
    const float* c1_wf   = (const float*)d_in[6];
    const float* c1_wd   = (const float*)d_in[7];
    const float* c1_g    = (const float*)d_in[8];
    const float* c1_b    = (const float*)d_in[9];
    const float* c2_wf   = (const float*)d_in[10];
    const float* c2_wd   = (const float*)d_in[11];
    const float* c2_g    = (const float*)d_in[12];
    const float* c2_b    = (const float*)d_in[13];
    const float* c3_w    = (const float*)d_in[14];
    const float* bn3_g   = (const float*)d_in[15];
    const float* bn3_b   = (const float*)d_in[16];
    const float* s1_wf   = (const float*)d_in[17];
    const float* s1_wd   = (const float*)d_in[18];
    const float* s1_g    = (const float*)d_in[19];
    const float* s1_b    = (const float*)d_in[20];
    const float* s2_wf   = (const float*)d_in[21];
    const float* s2_wd   = (const float*)d_in[22];
    const float* s2_g    = (const float*)d_in[23];
    const float* s2_b    = (const float*)d_in[24];
    const float* sl_w    = (const float*)d_in[25];
    float* out = (float*)d_out;

    char* ws = (char*)d_ws;
    size_t off = 0;
    auto alloc = [&](size_t bytes) -> void* {
        void* p = ws + off;
        off += (bytes + 255) & ~(size_t)255;
        return p;
    };

    int*       idx   = (int*)      alloc((size_t)B_ * NK_ * 4);
    float*     stats = (float*)    alloc(2048 * 4);
    float*     hE    = (float*)    alloc((size_t)B_ * C1_ * 3 * NK_ * 4);
    float*     dotE  = (float*)    alloc((size_t)B_ * C1_ * NK_ * 4);
    float*     h0    = (float*)    alloc((size_t)B_ * C1_ * NC_ * 4);
    float*     P1    = (float*)    alloc((size_t)B_ * C1_ * NC_ * 4);
    float*     D1    = (float*)    alloc((size_t)B_ * C1_ * NC_ * 4);
    float*     P2    = (float*)    alloc((size_t)B_ * C2_ * NC_ * 4);
    float*     D2    = (float*)    alloc((size_t)B_ * C2_ * NC_ * 4);
    _Float16*  Xp    = (_Float16*) alloc((size_t)B_ * 704 * NC_ * 2);    // max: s1 (KP=22)
    _Float16*  Wa    = (_Float16*) alloc((size_t)352 * 704 * 2);
    float*     P3    = (float*)    alloc((size_t)B_ * C3_ * NC_ * 4);    // -> h3
    float*     hmean = (float*)    alloc((size_t)B_ * C3_ * 3 * 4);
    float*     Ps1   = (float*)    alloc((size_t)B_ * C3_ * NC_ * 4);    // -> z1
    float*     Ds1   = (float*)    alloc((size_t)B_ * C3_ * NC_ * 4);
    float*     Ps2   = (float*)    alloc((size_t)B_ * Cq_ * NC_ * 4);    // -> z2
    float*     Ds2   = (float*)    alloc((size_t)B_ * Cq_ * NC_ * 4);
    float*     zfr   = (float*)    alloc((size_t)B_ * 9 * N_ * 4);

    const float invN = 1.0f / (float)(B_ * N_);

    // 1) kNN
    knn_kernel<<<dim3(N_ / 256, B_), 256, 0, stream>>>(x, idx);

    // 2) edge conv (cp) with batch-stat two-pass, then VN max-pool
    zero_kernel<<<8, 256, 0, stream>>>(stats, 2048);
    edge_stats_kernel<<<(B_ * NK_) / 256, 256, 0, stream>>>(x, idx, cp_wf, stats);
    edge_apply_kernel<<<(B_ * NK_) / 256, 256, 0, stream>>>(x, idx, cp_wf, cp_wd, cp_g, cp_b,
                                                            stats, pool_wd, hE, dotE);
    pool_kernel<<<divup(B_ * C1_ * N_, 256), 256, 0, stream>>>(hE, dotE, h0);

    // 3) c1 (21->21) VALU
    dual_linear_kernel<<<(B_ * NC_) / 256, 256, 0, stream>>>(h0, c1_wf, c1_wd, P1, D1, C1_, C1_);
    bn_stats_kernel<<<C1_, 256, 0, stream>>>(P1, stats, C1_);
    bn_apply_kernel<<<divup(B_ * C1_ * N_, 256), 256, 0, stream>>>(P1, D1, c1_g, c1_b, stats,
                                                                   P1, C1_, invN, 1);

    // 4) c2 (21->42) VALU
    dual_linear_kernel<<<(B_ * NC_) / 256, 256, 0, stream>>>(P1, c2_wf, c2_wd, P2, D2, C1_, C2_);
    bn_stats_kernel<<<C2_, 256, 0, stream>>>(P2, stats, C2_);
    bn_apply_kernel<<<divup(B_ * C2_ * N_, 256), 256, 0, stream>>>(P2, D2, c2_g, c2_b, stats,
                                                                   P2, C2_, invN, 1);

    // 5) c3 (42->341) WMMA GEMM (KP=2, MT=22) + VN-BN (no relu)
    pack_b_kernel<<<divup(B_ * 2 * NT_ * 512, 256), 256, 0, stream>>>(P2, Xp, C2_, 2);
    pack_a_kernel<<<divup(22 * 2 * 512, 256), 256, 0, stream>>>(c3_w, Wa, C3_, C2_, 2, 22);
    wmma_gemm_kernel<<<dim3(NT_ / 4, 22, B_), 32, 0, stream>>>(Wa, Xp, P3, C3_, 2, NC_);
    bn_stats_kernel<<<C3_, 256, 0, stream>>>(P3, stats, C3_);
    bn_apply_kernel<<<divup(B_ * C3_ * N_, 256), 256, 0, stream>>>(P3, nullptr, bn3_g, bn3_b,
                                                                   stats, P3, C3_, invN, 0);

    // 6) mean + concat to 682 channels, packed directly into fragment layout
    mean_kernel<<<divup(B_ * C3_ * 3, 256), 256, 0, stream>>>(P3, hmean);
    pack_b682_kernel<<<divup(B_ * 22 * NT_ * 512, 256), 256, 0, stream>>>(P3, hmean, Xp, 22);

    // 7) s1 (682->341) two WMMA GEMMs (KP=22, MT=22) + VN-BN + lrelu
    pack_a_kernel<<<divup(22 * 22 * 512, 256), 256, 0, stream>>>(s1_wf, Wa, C3_, C6_, 22, 22);
    wmma_gemm_kernel<<<dim3(NT_ / 4, 22, B_), 32, 0, stream>>>(Wa, Xp, Ps1, C3_, 22, NC_);
    pack_a_kernel<<<divup(22 * 22 * 512, 256), 256, 0, stream>>>(s1_wd, Wa, C3_, C6_, 22, 22);
    wmma_gemm_kernel<<<dim3(NT_ / 4, 22, B_), 32, 0, stream>>>(Wa, Xp, Ds1, C3_, 22, NC_);
    bn_stats_kernel<<<C3_, 256, 0, stream>>>(Ps1, stats, C3_);
    bn_apply_kernel<<<divup(B_ * C3_ * N_, 256), 256, 0, stream>>>(Ps1, Ds1, s1_g, s1_b, stats,
                                                                   Ps1, C3_, invN, 1);

    // 8) s2 (341->170) two WMMA GEMMs (KP=11, MT=11) + VN-BN + lrelu
    pack_b_kernel<<<divup(B_ * 11 * NT_ * 512, 256), 256, 0, stream>>>(Ps1, Xp, C3_, 11);
    pack_a_kernel<<<divup(11 * 11 * 512, 256), 256, 0, stream>>>(s2_wf, Wa, Cq_, C3_, 11, 11);
    wmma_gemm_kernel<<<dim3(NT_ / 4, 11, B_), 32, 0, stream>>>(Wa, Xp, Ps2, Cq_, 11, NC_);
    pack_a_kernel<<<divup(11 * 11 * 512, 256), 256, 0, stream>>>(s2_wd, Wa, Cq_, C3_, 11, 11);
    wmma_gemm_kernel<<<dim3(NT_ / 4, 11, B_), 32, 0, stream>>>(Wa, Xp, Ds2, Cq_, 11, NC_);
    bn_stats_kernel<<<Cq_, 256, 0, stream>>>(Ps2, stats, Cq_);
    bn_apply_kernel<<<divup(B_ * Cq_ * N_, 256), 256, 0, stream>>>(Ps2, Ds2, s2_g, s2_b, stats,
                                                                   Ps2, Cq_, invN, 1);

    // 9) frame linear (170->3) + final einsum + max over N
    sl_kernel<<<(B_ * NC_) / 256, 256, 0, stream>>>(Ps2, sl_w, zfr);
    final_kernel<<<divup(B_ * C6_ * 3, 256), 256, 0, stream>>>(P3, hmean, zfr, out);
}
